// TFS_72181220376979
// MI455X (gfx1250) — matrix operand order repaired and checked by reading, not compiled, not run on hardware
//
#include <hip/hip_runtime.h>

#define BB 32
#define PP 2048
#define DD 256

typedef __attribute__((ext_vector_type(16))) __bf16 v16bf;
typedef __attribute__((ext_vector_type(8)))  __bf16 v8bf;
typedef __attribute__((ext_vector_type(8)))  float  v8f;

// ---- WMMA fragment helpers (layouts per CDNA5 ISA 7.12.2, wave32) ----------
// A-matrix 16x32 bf16: lane L -> row m = L%16; half = L/16.
//   frag[i]   = A[m][k0 + half*8 + i]        (i = 0..7)
//   frag[i+8] = A[m][k0 + 16 + half*8 + i]
__device__ __forceinline__ v16bf load_a_frag_g(const __bf16* __restrict__ row,
                                               int k0, int half) {
  const v8bf c0 = *(const v8bf*)(row + k0 + half * 8);
  const v8bf c1 = *(const v8bf*)(row + k0 + 16 + half * 8);
  v16bf r;
#pragma unroll
  for (int i = 0; i < 8; i++) { r[i] = c0[i]; r[i + 8] = c1[i]; }
  return r;
}

// B-matrix 32x16 bf16: lane L -> column nn = L%16; half = L/16.
//   frag[i] = B[k0 + half*8 + i][nn], frag[i+8] = B[k0 + 16 + half*8 + i][nn]
//   (B stored column-major, i.e. we pass a pointer to row nn of B^T).
__device__ __forceinline__ v16bf load_b_frag_g(const __bf16* __restrict__ colrow,
                                               int k0, int half) {
  const v8bf c0 = *(const v8bf*)(colrow + k0 + half * 8);
  const v8bf c1 = *(const v8bf*)(colrow + k0 + 16 + half * 8);
  v16bf r;
#pragma unroll
  for (int i = 0; i < 8; i++) { r[i] = c0[i]; r[i + 8] = c1[i]; }
  return r;
}

__device__ __forceinline__ v8f wmma_bf16(v16bf a, v16bf b, v8f c) {
  return __builtin_amdgcn_wmma_f32_16x16x32_bf16(false, a, false, b,
                                                 (short)0, c, false, false);
}

// ---- prep kernels ----------------------------------------------------------
__global__ void convert_x_kernel(const float* __restrict__ x,
                                 __bf16* __restrict__ xb,
                                 __bf16* __restrict__ xbT) {
  const long long total = (long long)BB * PP * DD;
  long long idx = (long long)blockIdx.x * blockDim.x + threadIdx.x;
  if (idx >= total) return;
  const int d = (int)(idx % DD);
  const long long bp = idx / DD;
  const int p = (int)(bp % PP);
  const int b = (int)(bp / PP);
  const __bf16 v = (__bf16)x[idx];
  xb[idx] = v;
  xbT[((size_t)b * DD + d) * PP + p] = v;
}

// w [rows][cols] row-major f32 -> wT [cols][rows] bf16
__global__ void transpose_w_kernel(const float* __restrict__ w,
                                   __bf16* __restrict__ wT,
                                   int rows, int cols) {
  const long long total = (long long)rows * cols;
  long long idx = (long long)blockIdx.x * blockDim.x + threadIdx.x;
  if (idx >= total) return;
  const int c = (int)(idx % cols);
  const int r = (int)(idx / cols);
  wT[(size_t)c * rows + r] = (__bf16)w[idx];
}

// g2[k] = sum_d cur_w[k][d]*gate_w[d];  g2[DD] = cur_b.gate_w + gate_b
__global__ void compute_g2_kernel(const float* __restrict__ cur_w,
                                  const float* __restrict__ cur_b,
                                  const float* __restrict__ gate_w,
                                  const float* __restrict__ gate_b,
                                  float* __restrict__ g2) {
  const int k = threadIdx.x;
  float s = 0.f;
  for (int d = 0; d < DD; d++) s += cur_w[k * DD + d] * gate_w[d];
  g2[k] = s;
  if (k == 0) {
    float c0 = gate_b[0];
    for (int d = 0; d < DD; d++) c0 += cur_b[d] * gate_w[d];
    g2[DD] = c0;
  }
}

// ---- H = x @ his_w + his_b  (bf16 out) -------------------------------------
__global__ __launch_bounds__(32) void proj_kernel(const __bf16* __restrict__ xb,
                                                  const __bf16* __restrict__ hisT,
                                                  const float* __restrict__ his_b,
                                                  __bf16* __restrict__ Hb) {
  const int lane = threadIdx.x & 31;
  const int nn = lane & 15, half = lane >> 4;
  const int b = blockIdx.y;
  const int p0 = blockIdx.x * 16;
  const __bf16* xr = xb + ((size_t)b * PP + p0) * DD + (size_t)nn * DD;

  v16bf Ax[8];
#pragma unroll
  for (int kc = 0; kc < 8; kc++) Ax[kc] = load_a_frag_g(xr, kc * 32, half);

  for (int d = 0; d < 16; d++) {
    v8f h = {0, 0, 0, 0, 0, 0, 0, 0};
    const __bf16* wr = hisT + (size_t)(d * 16 + nn) * DD;
#pragma unroll
    for (int kc = 0; kc < 8; kc++)
      h = wmma_bf16(Ax[kc], load_b_frag_g(wr, kc * 32, half), h);
    const float bias = his_b[d * 16 + nn];
#pragma unroll
    for (int r = 0; r < 8; r++)  // C layout: row r+8*half, col nn
      Hb[((size_t)b * PP + p0 + r + 8 * half) * DD + d * 16 + nn] =
          (__bf16)(h[r] + bias);
  }
}

// ---- fused: logits -> softmax(full row) -> tril mask -> @x -> @mlp_w -> LN -
__global__ __launch_bounds__(32) void flash_kernel(
    const __bf16* __restrict__ Hb,    // [B][P][D]
    const __bf16* __restrict__ xbT,   // [B][D][P]
    const float* __restrict__ x,      // [B][P][D] f32
    const __bf16* __restrict__ wlT,   // [P][D]
    const float* __restrict__ wl_b,   // [P]
    const __bf16* __restrict__ mlpT,  // [D][D]
    const float* __restrict__ mlp_b,  // [D]
    const float* __restrict__ g2,     // [D+1]
    const float* __restrict__ ln_g,
    const float* __restrict__ ln_b,
    float* __restrict__ out) {        // [B][P][D]
  const int lane = threadIdx.x & 31;
  const int nn = lane & 15, half = lane >> 4;
  const int b = blockIdx.y;
  const int p0 = blockIdx.x * 16;

  __shared__ float Sls[16][32];
  __shared__ __bf16 Pls[16][32];
  __shared__ float m_run[16], l_run[16], rowscale[16];
  __shared__ __bf16 Aggls[16][DD];
  __shared__ float Hhls[16][DD];

  const __bf16* Hrow = Hb + ((size_t)b * PP + p0) * DD + (size_t)nn * DD;
  const __bf16* xT = xbT + (size_t)b * DD * PP;

  v16bf Ah[8];
#pragma unroll
  for (int kc = 0; kc < 8; kc++) Ah[kc] = load_a_frag_g(Hrow, kc * 32, half);

  const v8f vzero = {0, 0, 0, 0, 0, 0, 0, 0};
  v8f acc[16];
#pragma unroll
  for (int d = 0; d < 16; d++) acc[d] = vzero;

  if (lane < 16) { m_run[lane] = -1e30f; l_run[lane] = 0.f; }
  __syncthreads();

  for (int q0 = 0; q0 < PP; q0 += 32) {
    // --- S = H(tile) @ wl_w[:, q0:q0+32] + wl_b, two 16-col subtiles ---
#pragma unroll
    for (int t = 0; t < 2; t++) {
      const int q = q0 + t * 16;
      v8f s = vzero;
      const __bf16* wr = wlT + (size_t)(q + nn) * DD;
#pragma unroll
      for (int kc = 0; kc < 8; kc++)
        s = wmma_bf16(Ah[kc], load_b_frag_g(wr, kc * 32, half), s);
      const float bias = wl_b[q + nn];
#pragma unroll
      for (int r = 0; r < 8; r++) Sls[r + 8 * half][t * 16 + nn] = s[r] + bias;
    }
    __syncthreads();

    // --- online softmax: normalizer over ALL cols, causal mask on P only ---
    if (lane < 16) {
      const int row = lane, prow = p0 + row;
      float mloc = -1e30f;
#pragma unroll
      for (int c = 0; c < 32; c++) mloc = fmaxf(mloc, Sls[row][c]);
      const float mold = m_run[row];
      const float mnew = fmaxf(mold, mloc);
      const float sc = __expf(mold - mnew);
      float ssum = 0.f;
#pragma unroll
      for (int c = 0; c < 32; c++) {
        const float e = __expf(Sls[row][c] - mnew);
        ssum += e;  // full-row softmax denominator (pre-mask)
        Pls[row][c] = (__bf16)(((q0 + c) <= prow) ? e : 0.f);  // tril
      }
      l_run[row] = l_run[row] * sc + ssum;
      m_run[row] = mnew;
      rowscale[row] = sc;
    }
    __syncthreads();

    float scr[8];
#pragma unroll
    for (int r = 0; r < 8; r++) scr[r] = rowscale[r + 8 * half];

    v16bf Ap;  // P tile as A-fragment from LDS
#pragma unroll
    for (int i = 0; i < 8; i++) {
      Ap[i] = Pls[nn][half * 8 + i];
      Ap[i + 8] = Pls[nn][16 + half * 8 + i];
    }

    // --- acc(16xD) = acc*scale + P(16x32) @ x[q0:q0+32][:] ---
    for (int d = 0; d < 16; d++) {
      const __bf16* xr = xT + (size_t)(d * 16 + nn) * PP + q0;
      const v16bf bx = load_b_frag_g(xr, 0, half);
#pragma unroll
      for (int r = 0; r < 8; r++) acc[d][r] *= scr[r];
      acc[d] = wmma_bf16(Ap, bx, acc[d]);
    }
    __syncthreads();
  }

  // --- normalize: aggregated = acc / l_run, stage to LDS in bf16 ---
  float lr[8];
#pragma unroll
  for (int r = 0; r < 8; r++) lr[r] = 1.f / l_run[r + 8 * half];
#pragma unroll
  for (int d = 0; d < 16; d++)
#pragma unroll
    for (int r = 0; r < 8; r++)
      Aggls[r + 8 * half][d * 16 + nn] = (__bf16)(acc[d][r] * lr[r]);
  __syncthreads();

  // --- H_history = aggregated @ mlp_w + mlp_b (WMMA, A from LDS) ---
  v16bf Aa[8];
#pragma unroll
  for (int kc = 0; kc < 8; kc++)
#pragma unroll
    for (int i = 0; i < 8; i++) {
      Aa[kc][i] = Aggls[nn][kc * 32 + half * 8 + i];
      Aa[kc][i + 8] = Aggls[nn][kc * 32 + 16 + half * 8 + i];
    }
  for (int d = 0; d < 16; d++) {
    v8f h = vzero;
    const __bf16* wr = mlpT + (size_t)(d * 16 + nn) * DD;
#pragma unroll
    for (int kc = 0; kc < 8; kc++)
      h = wmma_bf16(Aa[kc], load_b_frag_g(wr, kc * 32, half), h);
    const float bias = mlp_b[d * 16 + nn];
#pragma unroll
    for (int r = 0; r < 8; r++) Hhls[r + 8 * half][d * 16 + nn] = h[r] + bias;
  }
  __syncthreads();

  // --- gate term + LayerNorm epilogue, one row per lane (lanes 0..15) ---
  if (lane < 16) {
    const int row = lane;
    const float* xr = x + ((size_t)b * PP + p0 + row) * DD;
    float wc = g2[DD];  // folded gate: wc = x . (cur_w@gate_w) + c0
    for (int k = 0; k < DD; k++) wc += xr[k] * g2[k];
    float mu = 0.f;
    for (int c = 0; c < DD; c++) {
      const float v = Hhls[row][c] + wc * xr[c];
      Hhls[row][c] = v;
      mu += v;
    }
    mu *= (1.f / DD);
    float var = 0.f;
    for (int c = 0; c < DD; c++) {
      const float dv = Hhls[row][c] - mu;
      var += dv * dv;
    }
    var *= (1.f / DD);
    const float rs = rsqrtf(var + 1e-5f);
    float* orow = out + ((size_t)b * PP + p0 + row) * DD;
    for (int c = 0; c < DD; c++)
      orow[c] = (Hhls[row][c] - mu) * rs * ln_g[c] + ln_b[c];
  }
}

// ---- host launcher ---------------------------------------------------------
extern "C" void kernel_launch(void* const* d_in, const int* in_sizes, int n_in,
                              void* d_out, int out_size, void* d_ws,
                              size_t ws_size, hipStream_t stream) {
  const float* x = (const float*)d_in[0];
  const float* mlp_w = (const float*)d_in[1];
  const float* mlp_b = (const float*)d_in[2];
  const float* his_w = (const float*)d_in[3];
  const float* his_b = (const float*)d_in[4];
  const float* cur_w = (const float*)d_in[5];
  const float* cur_b = (const float*)d_in[6];
  const float* wl_w = (const float*)d_in[7];
  const float* wl_b = (const float*)d_in[8];
  const float* gate_w = (const float*)d_in[9];
  const float* gate_b = (const float*)d_in[10];
  const float* ln_g = (const float*)d_in[11];
  const float* ln_b = (const float*)d_in[12];

  char* ws = (char*)d_ws;
  size_t off = 0;
  auto walloc = [&](size_t bytes) -> void* {
    void* p = ws + off;
    off = (off + bytes + 255) & ~((size_t)255);
    return p;
  };
  __bf16* xb = (__bf16*)walloc((size_t)BB * PP * DD * 2);
  __bf16* xbT = (__bf16*)walloc((size_t)BB * PP * DD * 2);
  __bf16* Hb = (__bf16*)walloc((size_t)BB * PP * DD * 2);
  __bf16* wlT = (__bf16*)walloc((size_t)PP * DD * 2);
  __bf16* hisT = (__bf16*)walloc((size_t)DD * DD * 2);
  __bf16* mlpT = (__bf16*)walloc((size_t)DD * DD * 2);
  float* g2 = (float*)walloc((size_t)(DD + 1) * 4);

  const long long total = (long long)BB * PP * DD;
  convert_x_kernel<<<dim3((unsigned)((total + 255) / 256)), dim3(256), 0,
                     stream>>>(x, xb, xbT);
  transpose_w_kernel<<<dim3((DD * PP + 255) / 256), dim3(256), 0, stream>>>(
      wl_w, wlT, DD, PP);
  transpose_w_kernel<<<dim3((DD * DD + 255) / 256), dim3(256), 0, stream>>>(
      his_w, hisT, DD, DD);
  transpose_w_kernel<<<dim3((DD * DD + 255) / 256), dim3(256), 0, stream>>>(
      mlp_w, mlpT, DD, DD);
  compute_g2_kernel<<<1, DD, 0, stream>>>(cur_w, cur_b, gate_w, gate_b, g2);

  proj_kernel<<<dim3(PP / 16, BB), dim3(32), 0, stream>>>(xb, hisT, his_b, Hb);

  flash_kernel<<<dim3(PP / 16, BB), dim3(32), 0, stream>>>(
      Hb, xbT, x, wlT, wl_b, mlpT, mlp_b, g2, ln_g, ln_b, (float*)d_out);

  // second tuple output: origin = x, straight device copy
  hipMemcpyAsync((float*)d_out + (size_t)BB * PP * DD, d_in[0],
                 (size_t)BB * PP * DD * sizeof(float),
                 hipMemcpyDeviceToDevice, stream);
}
